// kn_pr_Conv1D_34514357190888
// MI455X (gfx1250) — matrix operand order repaired and checked by reading, not compiled, not run on hardware
//
#include <hip/hip_runtime.h>

// Problem constants (from the reference):
//   R=4, M1=1024, N1=2048, M2=2, N2=4, BATCH=4, SEQ=2048
//   tokens = 8192, K = M1*M2 = 2048, N = N1*N2 = 8192
#define TOKENS 8192
#define KDIM   2048
#define NDIM   8192
#define R_     4
#define M1_    1024
#define N1_    2048
#define M2_    2
#define N2_    4

typedef __attribute__((ext_vector_type(16))) __bf16 v16bf;
typedef __attribute__((ext_vector_type(8)))  __bf16 v8bf;
typedef __attribute__((ext_vector_type(8)))  float  v8f;

__device__ __forceinline__ v16bf pack16(v8bf lo, v8bf hi) {
  v16bf r;
#pragma unroll
  for (int e = 0; e < 8; ++e) { r[e] = lo[e]; r[e + 8] = hi[e]; }
  return r;
}

// ---------------------------------------------------------------------------
// Phase 1: split x (fp32, token-major, K contiguous) into bf16 hi/lo planes.
// hi = rne(x), lo = rne(x - hi)  =>  x ~= hi + lo with ~2^-16 relative error.
// ---------------------------------------------------------------------------
__global__ __launch_bounds__(256) void k_x_split(const float* __restrict__ x,
                                                 __bf16* __restrict__ Xhi,
                                                 __bf16* __restrict__ Xlo) {
  const size_t idx = ((size_t)blockIdx.x * blockDim.x + threadIdx.x) * 4;
  const float4 v = *(const float4*)(x + idx);
  float f[4] = {v.x, v.y, v.z, v.w};
#pragma unroll
  for (int e = 0; e < 4; ++e) {
    const __bf16 hi = (__bf16)f[e];
    const __bf16 lo = (__bf16)(f[e] - (float)hi);
    Xhi[idx + e] = hi;
    Xlo[idx + e] = lo;
  }
}

// ---------------------------------------------------------------------------
// Phase 2: build the folded weight matrix, transposed (o-major) in bf16 hi/lo.
//   Wt[o][p_inv[i]] = sum_r A[r, q[o]>>2, i>>1] * B[r, i&1, q[o]&3]
// One block per output column o; writes to one 4KB row stay L2-resident.
// ---------------------------------------------------------------------------
__global__ __launch_bounds__(256) void k_w_build(const float* __restrict__ A,
                                                 const float* __restrict__ B,
                                                 const int* __restrict__ p_inv,
                                                 const int* __restrict__ q,
                                                 __bf16* __restrict__ Wthi,
                                                 __bf16* __restrict__ Wtlo) {
  const int o = blockIdx.x;
  const int c = q[o];
  const int n = c >> 2;
  const int j = c & 3;
  float Brm[R_][M2_];
#pragma unroll
  for (int r = 0; r < R_; ++r)
#pragma unroll
    for (int m2 = 0; m2 < M2_; ++m2)
      Brm[r][m2] = B[(r * M2_ + m2) * N2_ + j];

  for (int i = threadIdx.x; i < KDIM; i += 256) {
    const int m  = i >> 1;
    const int m2 = i & 1;
    float w = 0.f;
#pragma unroll
    for (int r = 0; r < R_; ++r)
      w = fmaf(A[((size_t)(r * N1_ + n)) * M1_ + m], Brm[r][m2], w);
    const __bf16 hi = (__bf16)w;
    const __bf16 lo = (__bf16)(w - (float)hi);
    const int f = p_inv[i];
    Wthi[(size_t)o * KDIM + f] = hi;
    Wtlo[(size_t)o * KDIM + f] = lo;
  }
}

// ---------------------------------------------------------------------------
// Phase 3: GEMM  out[t, o] = sum_f X[t,f] * Wt[o,f] + bias[o]
// Block tile 256x128 (8 waves, 4x2 wave grid); each wave a 64x64 macro-tile
// = 4x4 tiles of 16x16 f32 accumulators.
// Per K-step (32): 48 WMMAs (3-product compensated bf16: hi*hi+hi*lo+lo*hi)
// against 32 b128 loads -> 1.5 WMMA per load.
//
// Addressing: per-lane fragment offsets are loop-invariant u32 byte offsets
// (planes are 32 MB, so they fit in 32 bits); the K advance is a uniform
// base-pointer increment (SALU), keeping VALU — and the WMMA->VALU 4-coexec
// hazard NOPs — out of the inner loop.
//
// A-fragment (16-bit A 16x32, ISA 7.12.2): lane<16 -> M=lane, K in
// {0..7}U{16..23}; lane>=16 -> M=lane-16, K in {8..15}U{24..31}.
// B-fragment (16-bit B 32x16): N = lane&15; lane<16 -> K in {0..7}U{16..23},
// lane>=16 -> K in {8..15}U{24..31} (same K order as A).
// ---------------------------------------------------------------------------
__global__ __launch_bounds__(256) void k_gemm(const __bf16* __restrict__ Xhi,
                                              const __bf16* __restrict__ Xlo,
                                              const __bf16* __restrict__ Wthi,
                                              const __bf16* __restrict__ Wtlo,
                                              const float* __restrict__ bias,
                                              float* __restrict__ out) {
  const int lane    = threadIdx.x & 31;
  const int wid     = threadIdx.x >> 5;
  const int rowbase = blockIdx.y * 256 + (wid >> 1) * 64;  // token rows
  const int colbase = blockIdx.x * 128 + (wid & 1) * 64;   // output cols
  const int l15     = lane & 15;
  const int ka      = (lane < 16) ? 0 : 8;    // A K sub-offset per lane half
  const int kb      = (lane < 16) ? 0 : 8;    // B K sub-offset per lane half

  // Loop-invariant per-lane byte offsets into the K=0 column of each stream.
  unsigned aoff[4], boff[4];
#pragma unroll
  for (int rt = 0; rt < 4; ++rt)
    aoff[rt] = (unsigned)(((rowbase + rt * 16 + l15) * KDIM + ka) * 2);
#pragma unroll
  for (int ct = 0; ct < 4; ++ct)
    boff[ct] = (unsigned)(((colbase + ct * 16 + l15) * KDIM + kb) * 2);

  v8f acc[4][4];
#pragma unroll
  for (int rt = 0; rt < 4; ++rt)
#pragma unroll
    for (int ct = 0; ct < 4; ++ct)
#pragma unroll
      for (int v = 0; v < 8; ++v) acc[rt][ct][v] = 0.f;

  // Uniform base pointers, advanced 64 B (32 bf16 of K) per iteration.
  const char* pxh = (const char*)Xhi;
  const char* pxl = (const char*)Xlo;
  const char* pwh = (const char*)Wthi;
  const char* pwl = (const char*)Wtlo;

  for (int k0 = 0; k0 < KDIM; k0 += 32) {
    v16bf ahi[4], alo[4];
#pragma unroll
    for (int rt = 0; rt < 4; ++rt) {
      ahi[rt] = pack16(*(const v8bf*)(pxh + aoff[rt]),
                       *(const v8bf*)(pxh + aoff[rt] + 32));
      alo[rt] = pack16(*(const v8bf*)(pxl + aoff[rt]),
                       *(const v8bf*)(pxl + aoff[rt] + 32));
    }
    v16bf bhi[4], blo[4];
#pragma unroll
    for (int ct = 0; ct < 4; ++ct) {
      bhi[ct] = pack16(*(const v8bf*)(pwh + boff[ct]),
                       *(const v8bf*)(pwh + boff[ct] + 32));
      blo[ct] = pack16(*(const v8bf*)(pwl + boff[ct]),
                       *(const v8bf*)(pwl + boff[ct] + 32));
    }
#pragma unroll
    for (int rt = 0; rt < 4; ++rt)
#pragma unroll
      for (int ct = 0; ct < 4; ++ct) {
        acc[rt][ct] = __builtin_amdgcn_wmma_f32_16x16x32_bf16(
            false, ahi[rt], false, bhi[ct], (short)0, acc[rt][ct], false, false);
        acc[rt][ct] = __builtin_amdgcn_wmma_f32_16x16x32_bf16(
            false, ahi[rt], false, blo[ct], (short)0, acc[rt][ct], false, false);
        acc[rt][ct] = __builtin_amdgcn_wmma_f32_16x16x32_bf16(
            false, alo[rt], false, bhi[ct], (short)0, acc[rt][ct], false, false);
      }
    pxh += 64; pxl += 64; pwh += 64; pwl += 64;
  }

  // Epilogue: 16x16 f32 C/D layout — VGPR v: lanes 0-15 -> M=v, lanes 16-31 -> M=8+v.
  const int lrow = (lane < 16) ? 0 : 8;
#pragma unroll
  for (int rt = 0; rt < 4; ++rt)
#pragma unroll
    for (int ct = 0; ct < 4; ++ct) {
      const int col  = colbase + ct * 16 + l15;
      const float bb = bias[col];
      const int row0 = rowbase + rt * 16 + lrow;
#pragma unroll
      for (int v = 0; v < 8; ++v)
        out[(size_t)(row0 + v) * NDIM + col] = acc[rt][ct][v] + bb;
    }
}

// ---------------------------------------------------------------------------
// Launch: inputs in setup_inputs() order: x, A, B, b, p_inv, q.
// Workspace layout (128 MB): Xhi | Xlo | Wthi | Wtlo, 32 MB each.
// ---------------------------------------------------------------------------
extern "C" void kernel_launch(void* const* d_in, const int* in_sizes, int n_in,
                              void* d_out, int out_size, void* d_ws, size_t ws_size,
                              hipStream_t stream) {
  const float* x     = (const float*)d_in[0];
  const float* A     = (const float*)d_in[1];
  const float* B     = (const float*)d_in[2];
  const float* b     = (const float*)d_in[3];
  const int*   p_inv = (const int*)d_in[4];
  const int*   q     = (const int*)d_in[5];
  float* out = (float*)d_out;

  __bf16* Xhi  = (__bf16*)d_ws;
  __bf16* Xlo  = Xhi  + (size_t)TOKENS * KDIM;
  __bf16* Wthi = Xlo  + (size_t)TOKENS * KDIM;
  __bf16* Wtlo = Wthi + (size_t)NDIM   * KDIM;

  // Phase 1: split X into bf16 hi/lo planes (16.78M elems, 4/thread).
  k_x_split<<<(TOKENS * KDIM) / (256 * 4), 256, 0, stream>>>(x, Xhi, Xlo);

  // Phase 2: fold A, B, p_inv, q into transposed bf16 hi/lo weight planes.
  k_w_build<<<NDIM, 256, 0, stream>>>(A, B, p_inv, q, Wthi, Wtlo);

  // Phase 3: the GEMM (8192 x 2048 x 8192) + bias.
  dim3 grid(NDIM / 128, TOKENS / 256);
  k_gemm<<<grid, 256, 0, stream>>>(Xhi, Xlo, Wthi, Wtlo, b, out);
}
